// SwinTransformerBlock_78168404787407
// MI455X (gfx1250) — compile-verified, matrix operand order checked by reading, not run on hardware
//
#include <hip/hip_runtime.h>
#include <hip/hip_bf16.h>
#include <math.h>

// Swin block: B=32, H=W=56, C=384, NH=12, HD=32, WS=7, SS=3, N=49(pad 64), HID=1536
#define BB    32
#define HH    56
#define WWI   56
#define CC    384
#define NHH   12
#define HDD   32
#define SSS   3
#define NTOK  49
#define NP    64
#define NWIN  2048      // B * 64 windows
#define HIDD  1536

typedef _Float16 f16;
typedef _Float16 v8h  __attribute__((ext_vector_type(8)));
typedef _Float16 v16h __attribute__((ext_vector_type(16)));
typedef float    v8f  __attribute__((ext_vector_type(8)));

__device__ inline v8f vzero() {
    v8f z = {0.f,0.f,0.f,0.f,0.f,0.f,0.f,0.f};
    return z;
}

// ---- CDNA5 async global->LDS copy (ASYNCcnt-tracked, no VGPR round trip) ----
// VDST operand = per-lane LDS byte address (low 32 bits of the generic pointer
// to a __shared__ object are the LDS offset); VADDR = 64-bit global address.
__device__ inline void async_ld_b128(f16* lds_dst, const f16* g) {
    asm volatile("global_load_async_to_lds_b128 %0, %1, off"
                 :: "v"((unsigned)(uintptr_t)lds_dst), "v"(g)
                 : "memory");
}
__device__ inline void wait_async0() {
#if __has_builtin(__builtin_amdgcn_s_wait_asynccnt)
    __builtin_amdgcn_s_wait_asynccnt(0);
#else
    asm volatile("s_wait_asynccnt 0" ::: "memory");
#endif
}

// CDNA5 16-bit A/B fragment load (16x32 A or 32x16 B with B pre-transposed to [N][K]):
// lane 0-15 : row (lane&15), K = 0..7 and 16..23
// lane 16-31: row (lane&15), K = 8..15 and 24..31
__device__ inline v16h frag_ld(const f16* __restrict__ base, int ld, int lane) {
    const int r = lane & 15, sel = (lane >> 4) & 1;
    const f16* p = base + (size_t)r * ld + sel * 8;
    v8h lo = *(const v8h*)(p);
    v8h hi = *(const v8h*)(p + 16);
    v16h o;
#pragma unroll
    for (int i = 0; i < 8; ++i) { o[i] = lo[i]; o[i + 8] = hi[i]; }
    return o;
}

__device__ inline v8f wmma32(v16h a, v16h b, v8f c) {
    // D = A(16x32 f16) * B(32x16 f16) + C(16x16 f32)
    return __builtin_amdgcn_wmma_f32_16x16x32_f16(false, a, false, b, (short)0, c,
                                                  false, false);
}

// ---------------------------------------------------------------- weight prep
__global__ void k_transpose_w(const float* __restrict__ src, f16* __restrict__ dst,
                              int R, int Cc) {
    int i = blockIdx.x * blockDim.x + threadIdx.x;
    if (i >= R * Cc) return;
    int r = i / Cc, c = i % Cc;
    dst[(size_t)c * R + r] = (f16)src[i];
}

// ------------------------------------------- LN1 + roll(-3,-3) + window partition
__global__ __launch_bounds__(256) void k_ln_part(const float* __restrict__ x,
                                                 const float* __restrict__ g1,
                                                 const float* __restrict__ b1,
                                                 f16* __restrict__ hwin) {
    const int win = blockIdx.x;
    const int lane = threadIdx.x & 31, wid = threadIdx.x >> 5;
    const int b = win >> 6, wi = win & 63, wy = wi >> 3, wx = wi & 7;
    for (int t = wid; t < NP; t += 8) {
        f16* drow = hwin + ((size_t)win * NP + t) * CC;
        if (t >= NTOK) {                 // zero-pad rows 49..63
            for (int i = lane; i < CC; i += 32) drow[i] = (f16)0.f;
            continue;
        }
        const int ty = t / 7, tx = t % 7;
        const int h = (wy * 7 + ty + SSS) % HH;
        const int w = (wx * 7 + tx + SSS) % WWI;
        const float* srow = x + ((size_t)b * (HH * WWI) + h * WWI + w) * CC;
        float v[12];
        float s = 0.f;
#pragma unroll
        for (int i = 0; i < 12; ++i) { v[i] = srow[lane + 32 * i]; s += v[i]; }
#pragma unroll
        for (int mk = 1; mk < 32; mk <<= 1) s += __shfl_xor(s, mk, 32);
        const float mean = s * (1.f / CC);
        float var = 0.f;
#pragma unroll
        for (int i = 0; i < 12; ++i) { float d = v[i] - mean; var += d * d; }
#pragma unroll
        for (int mk = 1; mk < 32; mk <<= 1) var += __shfl_xor(var, mk, 32);
        const float inv = rsqrtf(var * (1.f / CC) + 1e-5f);
#pragma unroll
        for (int i = 0; i < 12; ++i) {
            int c = lane + 32 * i;
            drow[c] = (f16)((v[i] - mean) * inv * g1[c] + b1[c]);
        }
    }
}

// --------------------------------------------------- QKV GEMM (per window, 64x384)
// A tile (64x384 f16 = 48KB) staged in LDS via async copy; B weights stream
// from global (L2-resident).
__global__ __launch_bounds__(512) void k_qkv(const f16* __restrict__ hwin,
                                             const f16* __restrict__ wqkvT,
                                             const float* __restrict__ bqkv,
                                             f16* __restrict__ q,
                                             f16* __restrict__ k,
                                             f16* __restrict__ vT) {
    const int win = blockIdx.x;
    const int which = blockIdx.y;                 // 0=q 1=k 2=v
    const int lane = threadIdx.x & 31, wid = threadIdx.x >> 5;
    const int m0 = (wid & 3) * 16;
    const int ng = wid >> 2;                      // 0..3, each owns 6 N-tiles
    const int nloc = lane & 15, rofs = ((lane >> 4) & 1) * 8;

    __shared__ f16 sA[NP * CC];                   // 48 KB window activation tile
    {
        const f16* src = hwin + (size_t)win * NP * CC;
#pragma unroll
        for (int i = 0; i < 6; ++i) {
            const int e = (threadIdx.x + i * 512) * 8;
            async_ld_b128(sA + e, src + e);
        }
    }
    wait_async0();
    __syncthreads();

    v8f acc[6];
#pragma unroll
    for (int t = 0; t < 6; ++t) acc[t] = vzero();

    const f16* Bbase = wqkvT + ((size_t)which * CC + ng * 96) * CC;

    for (int k0 = 0; k0 < CC; k0 += 32) {
        v16h a = frag_ld(sA + (size_t)m0 * CC + k0, CC, lane);
#pragma unroll
        for (int t = 0; t < 6; ++t) {
            v16h bf = frag_ld(Bbase + (size_t)t * 16 * CC + k0, CC, lane);
            acc[t] = wmma32(a, bf, acc[t]);
        }
    }
#pragma unroll
    for (int t = 0; t < 6; ++t) {
        const int ncol = ng * 96 + t * 16 + nloc;     // 0..383 within q/k/v
        const int head = ncol >> 5, hd = ncol & 31;
        const float bias = bqkv[which * CC + ncol];
#pragma unroll
        for (int j = 0; j < 8; ++j) {
            const int tok = m0 + j + rofs;
            float val = acc[t][j] + bias;
            if (which == 0) {
                val *= 0.17677669529663688f;          // HD^-0.5
                q[(((size_t)win * NHH + head) * NP + tok) * HDD + hd] = (f16)val;
            } else if (which == 1) {
                k[(((size_t)win * NHH + head) * NP + tok) * HDD + hd] = (f16)val;
            } else {                                   // v stored transposed [hd][tok]
                vT[(((size_t)win * NHH + head) * HDD + hd) * NP + tok] = (f16)val;
            }
        }
    }
}

// -------------------- fused attention per (window, head): QK^T + bias/mask + softmax + PV
__global__ __launch_bounds__(128) void k_attn(const f16* __restrict__ q,
                                              const f16* __restrict__ kk,
                                              const f16* __restrict__ vT,
                                              const float* __restrict__ rpb,
                                              f16* __restrict__ awin) {
    const int head = blockIdx.x, win = blockIdx.y;
    const int lane = threadIdx.x & 31, wid = threadIdx.x >> 5;
    const int nloc = lane & 15, rofs = ((lane >> 4) & 1) * 8;
    const int m0 = wid * 16;
    __shared__ f16 sP[NP * NP];                   // 8 KB softmaxed scores

    const f16* qb = q  + (((size_t)win * NHH + head) * NP) * HDD;
    const f16* kb = kk + (((size_t)win * NHH + head) * NP) * HDD;
    const f16* vb = vT + (((size_t)win * NHH + head) * HDD) * NP;

    // S = q @ k^T  (M=64, K=32, N=64) -> 4 tiles per wave, single K step
    v16h aq = frag_ld(qb + (size_t)m0 * HDD, HDD, lane);
    v8f s[4];
#pragma unroll
    for (int nt = 0; nt < 4; ++nt) {
        v16h bk = frag_ld(kb + (size_t)nt * 16 * HDD, HDD, lane);
        s[nt] = wmma32(aq, bk, vzero());
    }

    const int wi = win & 63, wy = wi >> 3, wx = wi & 7;
#pragma unroll
    for (int j = 0; j < 8; ++j) {
        const int i = m0 + j + rofs;              // query token
        float v[4];
#pragma unroll
        for (int nt = 0; nt < 4; ++nt) {
            const int col = nt * 16 + nloc;       // key token
            float b;
            if (i >= NTOK || col >= NTOK) {
                b = -1e30f;
            } else {
                const int tyi = i / 7, txi = i % 7, tyj = col / 7, txj = col % 7;
                b = rpb[((tyi - tyj + 6) * 13 + (txi - txj + 6)) * NHH + head];
                const int hi = wy * 7 + tyi, hj = wy * 7 + tyj;
                const int wci = wx * 7 + txi, wcj = wx * 7 + txj;
                const int ri = (hi < 49 ? 0 : (hi < 53 ? 1 : 2)) * 3 +
                               (wci < 49 ? 0 : (wci < 53 ? 1 : 2));
                const int rj = (hj < 49 ? 0 : (hj < 53 ? 1 : 2)) * 3 +
                               (wcj < 49 ? 0 : (wcj < 53 ? 1 : 2));
                if (ri != rj) b -= 100.f;
            }
            v[nt] = s[nt][j] + b;
        }
        // row softmax: the 64 columns of row i live in one half-wave (4 vals/lane)
        float mx = fmaxf(fmaxf(v[0], v[1]), fmaxf(v[2], v[3]));
#pragma unroll
        for (int mk = 1; mk < 16; mk <<= 1) mx = fmaxf(mx, __shfl_xor(mx, mk, 32));
        float e[4], sum = 0.f;
#pragma unroll
        for (int nt = 0; nt < 4; ++nt) { e[nt] = __expf(v[nt] - mx); sum += e[nt]; }
#pragma unroll
        for (int mk = 1; mk < 16; mk <<= 1) sum += __shfl_xor(sum, mk, 32);
        const float inv = 1.f / sum;
#pragma unroll
        for (int nt = 0; nt < 4; ++nt)
            sP[i * NP + nt * 16 + nloc] = (f16)(e[nt] * inv);
    }
    __syncthreads();

    // out = P @ v  (M=64, K=64, N=32); B from vT -> contiguous lanes
    v8f o0 = vzero(), o1 = vzero();
    for (int k0 = 0; k0 < NP; k0 += 32) {
        v16h ap = frag_ld(sP + (size_t)m0 * NP + k0, NP, lane);
        o0 = wmma32(ap, frag_ld(vb + 0 * NP + k0, NP, lane), o0);
        o1 = wmma32(ap, frag_ld(vb + 16 * NP + k0, NP, lane), o1);
    }
    f16* arow = awin + (size_t)win * NP * CC;
#pragma unroll
    for (int j = 0; j < 8; ++j) {
        const int tok = m0 + j + rofs;
        arow[(size_t)tok * CC + head * HDD + nloc]      = (f16)o0[j];
        arow[(size_t)tok * CC + head * HDD + 16 + nloc] = (f16)o1[j];
    }
}

// ---------------------- proj GEMM + window reverse + unshift + residual (fp32 out)
__global__ __launch_bounds__(512) void k_proj(const f16* __restrict__ awin,
                                              const f16* __restrict__ wprojT,
                                              const float* __restrict__ bproj,
                                              const float* __restrict__ x,
                                              float* __restrict__ x1) {
    const int win = blockIdx.x;
    const int lane = threadIdx.x & 31, wid = threadIdx.x >> 5;
    const int m0 = (wid & 3) * 16, ng = wid >> 2;
    const int nloc = lane & 15, rofs = ((lane >> 4) & 1) * 8;

    __shared__ f16 sA[NP * CC];                   // 48 KB attention-output tile
    {
        const f16* src = awin + (size_t)win * NP * CC;
#pragma unroll
        for (int i = 0; i < 6; ++i) {
            const int e = (threadIdx.x + i * 512) * 8;
            async_ld_b128(sA + e, src + e);
        }
    }
    wait_async0();
    __syncthreads();

    v8f acc[6];
#pragma unroll
    for (int t = 0; t < 6; ++t) acc[t] = vzero();

    const f16* Bb = wprojT + (size_t)(ng * 96) * CC;
    for (int k0 = 0; k0 < CC; k0 += 32) {
        v16h a = frag_ld(sA + (size_t)m0 * CC + k0, CC, lane);
#pragma unroll
        for (int t = 0; t < 6; ++t)
            acc[t] = wmma32(a, frag_ld(Bb + (size_t)t * 16 * CC + k0, CC, lane), acc[t]);
    }
    const int b = win >> 6, wi = win & 63, wy = wi >> 3, wx = wi & 7;
#pragma unroll
    for (int j = 0; j < 8; ++j) {
        const int tok = m0 + j + rofs;
        if (tok >= NTOK) continue;                 // drop padded rows
        const int ty = tok / 7, tx = tok % 7;
        const int h = (wy * 7 + ty + SSS) % HH;
        const int w = (wx * 7 + tx + SSS) % WWI;
        const size_t row = ((size_t)b * (HH * WWI) + h * WWI + w) * CC;
#pragma unroll
        for (int t = 0; t < 6; ++t) {
            const int c = ng * 96 + t * 16 + nloc;
            x1[row + c] = x[row + c] + acc[t][j] + bproj[c];
        }
    }
}

// ------------------------------------------------------------------------- LN2
__global__ __launch_bounds__(256) void k_ln2(const float* __restrict__ x1,
                                             const float* __restrict__ g2,
                                             const float* __restrict__ b2,
                                             f16* __restrict__ xl) {
    const int lane = threadIdx.x & 31, wid = threadIdx.x >> 5;
    const size_t row = (size_t)blockIdx.x * 8 + wid;
    const float* s = x1 + row * CC;
    f16* d = xl + row * CC;
    float v[12];
    float sum = 0.f;
#pragma unroll
    for (int i = 0; i < 12; ++i) { v[i] = s[lane + 32 * i]; sum += v[i]; }
#pragma unroll
    for (int mk = 1; mk < 32; mk <<= 1) sum += __shfl_xor(sum, mk, 32);
    const float mean = sum * (1.f / CC);
    float var = 0.f;
#pragma unroll
    for (int i = 0; i < 12; ++i) { float dd = v[i] - mean; var += dd * dd; }
#pragma unroll
    for (int mk = 1; mk < 32; mk <<= 1) var += __shfl_xor(var, mk, 32);
    const float inv = rsqrtf(var * (1.f / CC) + 1e-5f);
#pragma unroll
    for (int i = 0; i < 12; ++i) {
        const int c = lane + 32 * i;
        d[c] = (f16)((v[i] - mean) * inv * g2[c] + b2[c]);
    }
}

// ---------- fused MLP: out = x1 + (gelu(xl@W1 + b1) @ W2 + b2), hidden chunked in LDS
__global__ __launch_bounds__(512) void k_mlp(const f16* __restrict__ xl,
                                             const f16* __restrict__ w1T,
                                             const float* __restrict__ bm1,
                                             const f16* __restrict__ w2T,
                                             const float* __restrict__ bm2,
                                             const float* __restrict__ x1,
                                             float* __restrict__ out) {
    const int lane = threadIdx.x & 31, wid = threadIdx.x >> 5;
    const int nloc = lane & 15, rofs = ((lane >> 4) & 1) * 8;
    const size_t row0 = (size_t)blockIdx.x * 64;
    __shared__ f16 sX[64 * CC];     // 48 KB activation tile
    __shared__ f16 sH[64 * 64];     //  8 KB hidden chunk

    {
        const f16* src = xl + row0 * CC;
#pragma unroll
        for (int i = 0; i < 6; ++i) {
            const int e = (threadIdx.x + i * 512) * 8;
            async_ld_b128(sX + e, src + e);
        }
    }
    wait_async0();
    __syncthreads();

    const int m0 = (wid & 3) * 16, ng = wid >> 2;
    v8f acc[6];
#pragma unroll
    for (int t = 0; t < 6; ++t) acc[t] = vzero();

    for (int ch = 0; ch < 24; ++ch) {             // 24 x 64-wide hidden chunks
        const int hb = ch * 64;
        // phase 1: each wave computes one hidden tile (m0, hb + ng*16)
        v8f hacc = vzero();
        const f16* Bb = w1T + (size_t)(hb + ng * 16) * CC;
        for (int k0 = 0; k0 < CC; k0 += 32) {
            v16h a = frag_ld(sX + (size_t)m0 * CC + k0, CC, lane);
            hacc = wmma32(a, frag_ld(Bb + k0, CC, lane), hacc);
        }
        __syncthreads();                           // prior chunk's sH reads done
        const int hcol = hb + ng * 16 + nloc;
        const float hbias = bm1[hcol];
#pragma unroll
        for (int j = 0; j < 8; ++j) {
            float v = hacc[j] + hbias;
            v = 0.5f * v * (1.f + erff(v * 0.70710678118654752f));   // exact gelu
            sH[(m0 + j + rofs) * 64 + ng * 16 + nloc] = (f16)v;
        }
        __syncthreads();
        // phase 2: out += Hc @ W2[hb:hb+64, :]
#pragma unroll
        for (int ks = 0; ks < 2; ++ks) {
            const int k0 = ks * 32;
            v16h a = frag_ld(sH + (size_t)m0 * 64 + k0, 64, lane);
#pragma unroll
            for (int t = 0; t < 6; ++t) {
                const f16* B2 = w2T + (size_t)(ng * 96 + t * 16) * HIDD + hb;
                acc[t] = wmma32(a, frag_ld(B2 + k0, HIDD, lane), acc[t]);
            }
        }
    }
#pragma unroll
    for (int t = 0; t < 6; ++t) {
        const int c = ng * 96 + t * 16 + nloc;
        const float bias = bm2[c];
#pragma unroll
        for (int j = 0; j < 8; ++j) {
            const size_t r = row0 + m0 + j + rofs;
            out[r * CC + c] = x1[r * CC + c] + acc[t][j] + bias;
        }
    }
}

// ------------------------------------------------------------------- launcher
extern "C" void kernel_launch(void* const* d_in, const int* in_sizes, int n_in,
                              void* d_out, int out_size, void* d_ws, size_t ws_size,
                              hipStream_t stream) {
    (void)in_sizes; (void)n_in; (void)out_size; (void)ws_size;
    const float* x     = (const float*)d_in[0];
    const float* g1    = (const float*)d_in[1];
    const float* b1    = (const float*)d_in[2];
    const float* wqkv  = (const float*)d_in[3];
    const float* bqkv  = (const float*)d_in[4];
    const float* wproj = (const float*)d_in[5];
    const float* bproj = (const float*)d_in[6];
    const float* rpb   = (const float*)d_in[7];
    const float* g2    = (const float*)d_in[8];
    const float* b2    = (const float*)d_in[9];
    const float* w1    = (const float*)d_in[10];
    const float* bm1   = (const float*)d_in[11];
    const float* w2    = (const float*)d_in[12];
    const float* bm2   = (const float*)d_in[13];
    float* out = (float*)d_out;

    char* ws = (char*)d_ws;
    size_t off = 0;
    auto carve = [&](size_t bytes) -> char* {
        char* p = ws + off;
        off += (bytes + 255) & ~(size_t)255;
        return p;
    };
    f16*   wqkvT  = (f16*)carve((size_t)3 * CC * CC * 2);
    f16*   wprojT = (f16*)carve((size_t)CC * CC * 2);
    f16*   w1T    = (f16*)carve((size_t)CC * HIDD * 2);
    f16*   w2T    = (f16*)carve((size_t)HIDD * CC * 2);
    f16*   hwin   = (f16*)carve((size_t)NWIN * NP * CC * 2);
    f16*   qb     = (f16*)carve((size_t)NWIN * NP * CC * 2);
    f16*   kb     = (f16*)carve((size_t)NWIN * NP * CC * 2);
    f16*   vTb    = (f16*)carve((size_t)NWIN * NP * CC * 2);
    f16*   aw     = (f16*)carve((size_t)NWIN * NP * CC * 2);
    float* x1     = (float*)carve((size_t)BB * HH * WWI * CC * 4);
    f16*   xlbuf  = (f16*)carve((size_t)BB * HH * WWI * CC * 2);

    // weight transposes (fp32 [K][N] -> f16 [N][K], WMMA-B friendly)
    {
        int n;
        n = CC * 3 * CC; k_transpose_w<<<(n + 255) / 256, 256, 0, stream>>>(wqkv, wqkvT, CC, 3 * CC);
        n = CC * CC;     k_transpose_w<<<(n + 255) / 256, 256, 0, stream>>>(wproj, wprojT, CC, CC);
        n = CC * HIDD;   k_transpose_w<<<(n + 255) / 256, 256, 0, stream>>>(w1, w1T, CC, HIDD);
        n = HIDD * CC;   k_transpose_w<<<(n + 255) / 256, 256, 0, stream>>>(w2, w2T, HIDD, CC);
    }
    k_ln_part<<<NWIN, 256, 0, stream>>>(x, g1, b1, hwin);
    k_qkv<<<dim3(NWIN, 3), 512, 0, stream>>>(hwin, wqkvT, bqkv, qb, kb, vTb);
    k_attn<<<dim3(NHH, NWIN), 128, 0, stream>>>(qb, kb, vTb, rpb, aw);
    k_proj<<<NWIN, 512, 0, stream>>>(aw, wprojT, bproj, x, x1);
    k_ln2<<<12544, 256, 0, stream>>>(x1, g2, b2, xlbuf);
    k_mlp<<<1568, 512, 0, stream>>>(xlbuf, w1T, bm1, w2T, bm2, x1, out);
}
